// MultiHeadAttention_24429773980057
// MI455X (gfx1250) — matrix operand order repaired and checked by reading, not compiled, not run on hardware
//
#include <hip/hip_runtime.h>

#define BATCH 2
#define SEQ   4096
#define EMBED 512
#define HEADS 8
#define DHEAD 64
#define BSQ   (BATCH * SEQ)
#define KT    32   // key tile size in attention

typedef __bf16 bf16;
typedef bf16  v16bf __attribute__((ext_vector_type(16)));
typedef bf16  v8bf  __attribute__((ext_vector_type(8)));
typedef bf16  v4bf  __attribute__((ext_vector_type(4)));
typedef float v8f   __attribute__((ext_vector_type(8)));
typedef int   i32x4 __attribute__((vector_size(16)));

#if defined(__gfx1250__) && \
    __has_builtin(__builtin_amdgcn_global_load_async_to_lds_b128) && \
    __has_builtin(__builtin_amdgcn_s_wait_asynccnt)
#define HAVE_ASYNC_LDS 1
#define async_wait_le(n) __builtin_amdgcn_s_wait_asynccnt(n)
#else
#define HAVE_ASYNC_LDS 0
#define async_wait_le(n) ((void)0)
#endif

static __device__ __forceinline__ bf16 f2bf(float f) {
  unsigned u = __builtin_bit_cast(unsigned, f);
  u += 0x7fffu + ((u >> 16) & 1u);              // round-to-nearest-even
  unsigned short h = (unsigned short)(u >> 16);
  return __builtin_bit_cast(bf16, h);
}

static __device__ __forceinline__ v16bf cat8(v8bf lo, v8bf hi) {
  return __builtin_shufflevector(lo, hi, 0,1,2,3,4,5,6,7,8,9,10,11,12,13,14,15);
}

static __device__ __forceinline__ v8f wmma_bf16(v16bf a, v16bf b, v8f c) {
  // D = A(16x32 bf16) * B(32x16 bf16) + C(16x16 f32)
  return __builtin_amdgcn_wmma_f32_16x16x32_bf16(false, a, false, b, (short)0, c,
                                                 false, false);
}

// 16-byte global -> LDS copy: async on CDNA5 (ASYNCcnt), sync fallback.
static __device__ __forceinline__ void copy16_g2l(const bf16* g, bf16* l) {
#if HAVE_ASYNC_LDS
  __builtin_amdgcn_global_load_async_to_lds_b128(
      (__attribute__((address_space(1))) i32x4*)g,
      (__attribute__((address_space(3))) i32x4*)l, 0, 0);
#else
  *(v8bf*)l = *(const v8bf*)g;
#endif
}

// ---------------------------------------------------------------------------
// f32 -> bf16 conversion (4 elements per thread)
// ---------------------------------------------------------------------------
__global__ void cvt_f32_to_bf16(const float* __restrict__ in,
                                bf16* __restrict__ out, int n4) {
  const int i = blockIdx.x * blockDim.x + threadIdx.x;
  if (i >= n4) return;
  const float4 f = ((const float4*)in)[i];
  v4bf o;
  o[0] = f2bf(f.x); o[1] = f2bf(f.y); o[2] = f2bf(f.z); o[3] = f2bf(f.w);
  ((v4bf*)out)[i] = o;
}

// ---------------------------------------------------------------------------
// GEMM: out[M,N] = A[M,K] * W[N,K]^T + bias   (torch Linear semantics)
// One wave computes a 16(M) x 64(N) tile; K-loop in steps of 32.
// mode 0: f32 out[m*N+col]
// mode 1: bf16 out[m*N+col]
// mode 2: bf16 transposed V store -> Vt[b,h,d,s]   (m=(b,s), col=(h,d))
// ---------------------------------------------------------------------------
__global__ void __launch_bounds__(32)
gemm_wmma_bf16(const bf16* __restrict__ A, const bf16* __restrict__ W,
               const float* __restrict__ bias,
               bf16* __restrict__ outb, float* __restrict__ outf,
               int M, int N, int K, int mode) {
  const int lane = threadIdx.x & 31;
  const int l16  = lane & 15;
  const int half = lane >> 4;
  const int mtiles = M >> 4;
  const int mt = blockIdx.x % mtiles;
  const int nt = blockIdx.x / mtiles;
  const int n0 = nt * 64;

  const long aBase = (long)(mt * 16 + l16) * K;
  v8f acc[4];
#pragma unroll
  for (int j = 0; j < 4; ++j)
#pragma unroll
    for (int r = 0; r < 8; ++r) acc[j][r] = 0.0f;

  for (int k0 = 0; k0 < K; k0 += 32) {
    // A fragment: lane holds row (l16); 16 contiguous K values half*16..+15
    const v8bf alo = *(const v8bf*)(A + aBase + k0 + half * 16);
    const v8bf ahi = *(const v8bf*)(A + aBase + k0 + half * 16 + 8);
    const v16bf a = cat8(alo, ahi);
#pragma unroll
    for (int j = 0; j < 4; ++j) {
      // B fragment: lane holds column (l16) of B = W^T; 16 contiguous K values
      const long wBase = (long)(n0 + j * 16 + l16) * K;
      const v16bf b = *(const v16bf*)(W + wBase + k0 + half * 16);
      acc[j] = wmma_bf16(a, b, acc[j]);
    }
  }

  const int mbase = mt * 16 + half * 8;  // C layout: VGPR r -> row mbase+r
#pragma unroll
  for (int j = 0; j < 4; ++j) {
    const int col = n0 + j * 16 + l16;
    const float bb = bias[col];
#pragma unroll
    for (int r = 0; r < 8; ++r) {
      const int m = mbase + r;
      const float val = acc[j][r] + bb;
      if (mode == 0) {
        outf[(long)m * N + col] = val;
      } else if (mode == 1) {
        outb[(long)m * N + col] = f2bf(val);
      } else {
        // fused transpose for V: Vt[((b*H + h)*D + d)*S + s]
        const int bI = m >> 12;            // m / SEQ
        const int sI = m & (SEQ - 1);
        const int hI = col >> 6;           // col / DHEAD
        const int dI = col & (DHEAD - 1);
        outb[(long)((bI * HEADS + hI) * DHEAD + dI) * SEQ + sI] = f2bf(val);
      }
    }
  }
}

// ---------------------------------------------------------------------------
// Flash attention, 4 waves / block, 64 queries / block, one (b,h) per block.
// K/V 32-key tiles are double-buffered in LDS via async global->LDS loads and
// shared by all 4 waves. scores = Q K^T / sqrt(EMBED); online softmax; O += PV.
// ---------------------------------------------------------------------------
__global__ void __launch_bounds__(128)
flash_attn_wmma(const bf16* __restrict__ Q, const bf16* __restrict__ Km,
                const bf16* __restrict__ Vt, bf16* __restrict__ Ctx) {
  __shared__ alignas(128) bf16 kT[2][KT][DHEAD];   // 8 KB: keys row-major
  __shared__ alignas(128) bf16 vT[2][DHEAD][KT];   // 8 KB: V^T (d-major)
  __shared__ alignas(64)  bf16 pT[4][16 * 32];     // 4 KB: per-wave P tiles

  const int tid  = threadIdx.x;
  const int wave = tid >> 5;
  const int lane = tid & 31;
  const int l16  = lane & 15;
  const int half = lane >> 4;

  const int qb = blockIdx.x & (SEQ / 64 - 1);
  const int h  = (blockIdx.x >> 6) & (HEADS - 1);
  const int b  = blockIdx.x >> 9;
  const int s0 = qb * 64 + wave * 16;
  const float scale = 0.044194173824159216f;  // 1/sqrt(512)

  const long kBase = (long)b * SEQ * EMBED + h * DHEAD;
  const long vBase = (long)((b * HEADS + h) * DHEAD) * SEQ;

  // Cooperative stage of one 32-key K tile (32x64) and V^T tile (64x32):
  // 4 async b128 issues per wave per tile-pair (128 lanes x 16B x 4 = 8 KB).
  auto loadTiles = [&](int j0, int buf) {
#pragma unroll
    for (int qq = 0; qq < 2; ++qq) {
      const int c = qq * 128 + tid;           // 0..255 16B-chunk index
      const int krow = c >> 3, kcol = (c & 7) * 8;
      copy16_g2l(Km + kBase + (long)(j0 + krow) * EMBED + kcol,
                 &kT[buf][krow][kcol]);
      const int vrow = c >> 2, vcol = (c & 3) * 8;
      copy16_g2l(Vt + vBase + (long)vrow * SEQ + j0 + vcol,
                 &vT[buf][vrow][vcol]);
    }
  };

  // Q A-fragments for the two 32-wide K-dim steps covering D=64
  const long qRow = (long)(b * SEQ + s0 + l16) * EMBED + h * DHEAD;
  v16bf qa[2];
#pragma unroll
  for (int t = 0; t < 2; ++t) {
    const v8bf lo = *(const v8bf*)(Q + qRow + t * 32 + half * 16);
    const v8bf hi = *(const v8bf*)(Q + qRow + t * 32 + half * 16 + 8);
    qa[t] = cat8(lo, hi);
  }

  float rowMax[8], rowSum[8];
  v8f o[4];
#pragma unroll
  for (int r = 0; r < 8; ++r) { rowMax[r] = -3.0e38f; rowSum[r] = 0.0f; }
#pragma unroll
  for (int n = 0; n < 4; ++n)
#pragma unroll
    for (int r = 0; r < 8; ++r) o[n][r] = 0.0f;

  loadTiles(0, 0);

  for (int j0 = 0; j0 < SEQ; j0 += KT) {
    const int buf = (j0 >> 5) & 1;
    if (j0 + KT < SEQ) {
      loadTiles(j0 + KT, buf ^ 1);  // prefetch next tile into other buffer
      async_wait_le(4);             // this tile's 4 async loads have landed
    } else {
      async_wait_le(0);
    }
    __syncthreads();                // all waves' pieces of this tile visible

    // ---- scores: two 16x16 tiles from LDS K tile ----
    v8f e[2];
#pragma unroll
    for (int t = 0; t < 2; ++t) {
#pragma unroll
      for (int r = 0; r < 8; ++r) e[t][r] = 0.0f;
#pragma unroll
      for (int kk = 0; kk < 2; ++kk) {
        const v16bf kb =
            *(const v16bf*)(&kT[buf][t * 16 + l16][kk * 32 + half * 16]);
        e[t] = wmma_bf16(qa[kk], kb, e[t]);
      }
    }

    // ---- online softmax (per-row; a row lives in one 16-lane half) ----
#pragma unroll
    for (int r = 0; r < 8; ++r) {
      const float s0v = e[0][r] * scale;
      const float s1v = e[1][r] * scale;
      float mx = fmaxf(s0v, s1v);
      mx = fmaxf(mx, __shfl_xor(mx, 1, 32));
      mx = fmaxf(mx, __shfl_xor(mx, 2, 32));
      mx = fmaxf(mx, __shfl_xor(mx, 4, 32));
      mx = fmaxf(mx, __shfl_xor(mx, 8, 32));
      const float nm   = fmaxf(rowMax[r], mx);
      const float corr = __expf(rowMax[r] - nm);
      rowMax[r] = nm;
      const float p0 = __expf(s0v - nm);
      const float p1 = __expf(s1v - nm);
      float ts = p0 + p1;
      ts += __shfl_xor(ts, 1, 32);
      ts += __shfl_xor(ts, 2, 32);
      ts += __shfl_xor(ts, 4, 32);
      ts += __shfl_xor(ts, 8, 32);
      rowSum[r] = rowSum[r] * corr + ts;
#pragma unroll
      for (int n = 0; n < 4; ++n) o[n][r] *= corr;
      const int m = r + half * 8;
      pT[wave][m * 32 + l16]      = f2bf(p0);  // C-layout -> row-major LDS
      pT[wave][m * 32 + 16 + l16] = f2bf(p1);
    }
    // (per-wave P buffer: in-wave DS ordering suffices, no barrier needed)
    const v8bf plo = *(const v8bf*)(&pT[wave][l16 * 32 + half * 16]);
    const v8bf phi = *(const v8bf*)(&pT[wave][l16 * 32 + half * 16 + 8]);
    const v16bf pf = cat8(plo, phi);

    // ---- O += P * V : 4 n-subtiles of 16 over D=64, V from LDS ----
#pragma unroll
    for (int n = 0; n < 4; ++n) {
      const v16bf vb = *(const v16bf*)(&vT[buf][n * 16 + l16][half * 16]);
      o[n] = wmma_bf16(pf, vb, o[n]);
    }
    __syncthreads();  // done reading this tile before it is overwritten
  }

  // finalize: O /= rowSum, store context (bf16) back in [B,S,H,D] layout
#pragma unroll
  for (int n = 0; n < 4; ++n) {
#pragma unroll
    for (int r = 0; r < 8; ++r) {
      const int m = r + half * 8;
      const float val = o[n][r] / rowSum[r];
      Ctx[(long)(b * SEQ + s0 + m) * EMBED + h * DHEAD + n * 16 + l16] = f2bf(val);
    }
  }
}

// ---------------------------------------------------------------------------
extern "C" void kernel_launch(void* const* d_in, const int* in_sizes, int n_in,
                              void* d_out, int out_size, void* d_ws, size_t ws_size,
                              hipStream_t stream) {
  (void)in_sizes; (void)n_in; (void)out_size; (void)ws_size;
  const float* x  = (const float*)d_in[0];
  const float* Wv = (const float*)d_in[1];
  const float* bv = (const float*)d_in[2];
  const float* Wk = (const float*)d_in[3];
  const float* bk = (const float*)d_in[4];
  const float* Wq = (const float*)d_in[5];
  const float* bq = (const float*)d_in[6];
  const float* Wo = (const float*)d_in[7];
  const float* bo = (const float*)d_in[8];
  float* out = (float*)d_out;

  char* ws = (char*)d_ws;
  size_t off = 0;
  auto wsAlloc = [&](size_t bytes) -> void* {
    void* p = ws + off;
    off += (bytes + 255) & ~(size_t)255;
    return p;
  };
  const size_t actBytes = (size_t)BSQ * EMBED * sizeof(bf16);   // 8 MB
  const size_t wBytes   = (size_t)EMBED * EMBED * sizeof(bf16); // 0.5 MB
  bf16* xb  = (bf16*)wsAlloc(actBytes);
  bf16* wqb = (bf16*)wsAlloc(wBytes);
  bf16* wkb = (bf16*)wsAlloc(wBytes);
  bf16* wvb = (bf16*)wsAlloc(wBytes);
  bf16* wob = (bf16*)wsAlloc(wBytes);
  bf16* q   = (bf16*)wsAlloc(actBytes);
  bf16* k   = (bf16*)wsAlloc(actBytes);
  bf16* vt  = (bf16*)wsAlloc(actBytes);   // V stored pre-transposed [B,H,D,S]
  bf16* ctx = (bf16*)wsAlloc(actBytes);

  const int nAct4 = BSQ * EMBED / 4;
  const int nW4   = EMBED * EMBED / 4;
  cvt_f32_to_bf16<<<nAct4 / 256, 256, 0, stream>>>(x, xb, nAct4);
  cvt_f32_to_bf16<<<nW4 / 256, 256, 0, stream>>>(Wq, wqb, nW4);
  cvt_f32_to_bf16<<<nW4 / 256, 256, 0, stream>>>(Wk, wkb, nW4);
  cvt_f32_to_bf16<<<nW4 / 256, 256, 0, stream>>>(Wv, wvb, nW4);
  cvt_f32_to_bf16<<<nW4 / 256, 256, 0, stream>>>(Wo, wob, nW4);

  const int gemmBlocks = (BSQ / 16) * (EMBED / 64);  // 4096 waves
  gemm_wmma_bf16<<<gemmBlocks, 32, 0, stream>>>(xb, wqb, bq, q,  nullptr, BSQ, EMBED, EMBED, 1);
  gemm_wmma_bf16<<<gemmBlocks, 32, 0, stream>>>(xb, wkb, bk, k,  nullptr, BSQ, EMBED, EMBED, 1);
  gemm_wmma_bf16<<<gemmBlocks, 32, 0, stream>>>(xb, wvb, bv, vt, nullptr, BSQ, EMBED, EMBED, 2);

  flash_attn_wmma<<<BATCH * HEADS * (SEQ / 64), 128, 0, stream>>>(q, k, vt, ctx);

  gemm_wmma_bf16<<<gemmBlocks, 32, 0, stream>>>(ctx, wob, bo, nullptr, out, BSQ, EMBED, EMBED, 0);
}
